// MultiHeadAttention_54692113548034
// MI455X (gfx1250) — matrix operand order repaired and checked by reading, not compiled, not run on hardware
//
#include <hip/hip_runtime.h>
#include <hip/hip_bf16.h>

#define B_ 128
#define T_ 256
#define C_ 384
#define H_ 6
#define HS_ 64
#define F3_ 1152          // 3*H*HS
#define M_ (B_*T_)        // 32768

typedef __attribute__((ext_vector_type(16))) __bf16 v16bf;
typedef __attribute__((ext_vector_type(8)))  float  v8f;
typedef unsigned short u16;
typedef unsigned int   u32;

static __device__ __forceinline__ u16 f2bf(float f){
  union { float f; u32 u; } x; x.f = f;
  u32 r = (x.u + 0x7FFFu + ((x.u >> 16) & 1u)) >> 16;
  return (u16)r;
}

static __device__ __forceinline__ v8f wmma_bf16(v16bf a, v16bf b, v8f c){
  // D = A(16x32) * B(32x16) + C, fp32 accumulate
  return __builtin_amdgcn_wmma_f32_16x16x32_bf16(false, a, false, b, (short)0, c, false, false);
}

// A fragment (16x32 bf16), lane: m = lane%16, g = lane/16.
// ISA layout: VGPR0-3 hold K=8g..8g+7, VGPR4-7 hold K=16+8g..16+8g+7.
static __device__ __forceinline__ v16bf load_a(const u16* rowk0, int g){
  v16bf r;
  ((uint4*)&r)[0] = *(const uint4*)(rowk0 + 8*g);
  ((uint4*)&r)[1] = *(const uint4*)(rowk0 + 16 + 8*g);
  return r;
}
// B fragment (32x16 bf16), lane: n = lane%16, g = lane/16; K = 8g..8g+7, 16+8g..16+8g+7.
static __device__ __forceinline__ v16bf load_b(const u16* rowk0, int g){
  v16bf r;
  ((uint4*)&r)[0] = *(const uint4*)(rowk0 + 8*g);
  ((uint4*)&r)[1] = *(const uint4*)(rowk0 + 16 + 8*g);
  return r;
}

// 32-byte global -> LDS async copy (tracked by ASYNCcnt, no dest VGPRs).
// INST_OFFSET is added to BOTH the LDS and the global address (ISA 10.7.4).
static __device__ __forceinline__ void async_copy32(u32 lds_off, const u16* g){
  asm volatile("global_load_async_to_lds_b128 %0, %1, off\n\t"
               "global_load_async_to_lds_b128 %0, %1, off offset:16"
               :: "v"(lds_off), "v"(g) : "memory");
}
static __device__ __forceinline__ void wait_async0(){
  asm volatile("s_wait_asynccnt 0" ::: "memory");
}

__global__ void cvt_f32_bf16(const float* __restrict__ in, u16* __restrict__ out, int n){
  int i = blockIdx.x * 256 + threadIdx.x;
  if (i < n) out[i] = f2bf(in[i]);
}

// C[m][n] = sum_k A[m][k]*B[n][k] (+bias[n]); A: MxK bf16, B: NxK bf16.
// 256 threads = 8 waves; block tile 128x128; wave tile 32x64.
// Double-buffered async global->LDS staging; all 6 fragments loaded up front
// per K-step so the 8 WMMAs issue back-to-back behind a single DScnt wait.
__global__ __launch_bounds__(256) void gemm_bf16_wmma(
    const u16* __restrict__ A, const u16* __restrict__ Bw,
    const float* __restrict__ bias, void* __restrict__ Cout,
    int M, int N, int K, int store_f32)
{
  __shared__ u16 As[2][128*32];   // 2 x 8 KB
  __shared__ u16 Bs[2][128*32];   // 2 x 8 KB

  const int tid  = threadIdx.x;
  const int lane = tid & 31, wave = tid >> 5;
  const int lr = lane & 15, lg = lane >> 4;
  const int mblk = blockIdx.x * 128;
  const int nblk = blockIdx.y * 128;
  const int wm = (wave & 3) * 32;     // wave's row offset inside block tile
  const int wn = (wave >> 2) * 64;    // wave's col offset inside block tile

  // Cooperative async staging: thread copies 32B of one A row and one B row.
  const int crow = tid >> 1;              // 0..127
  const int cofs = (tid & 1) * 16;        // element offset within 32-elem K slab
  const u16* gA = A  + (size_t)(mblk + crow) * K + cofs;
  const u16* gB = Bw + (size_t)(nblk + crow) * K + cofs;
  const u32 ldsA = (u32)(size_t)&As[0][0] + (u32)(crow*64 + cofs*2);
  const u32 ldsB = (u32)(size_t)&Bs[0][0] + (u32)(crow*64 + cofs*2);

  v8f zero = {};
  v8f acc[2][4];
#pragma unroll
  for (int i = 0; i < 2; ++i)
#pragma unroll
    for (int j = 0; j < 4; ++j) acc[i][j] = zero;

  const int nk = K / 32;
  // prologue: fill stage 0
  async_copy32(ldsA, gA);
  async_copy32(ldsB, gB);

  for (int it = 0; it < nk; ++it){
    const int cur = it & 1;
    wait_async0();          // own async stores for stage `cur` complete
    __syncthreads();        // whole stage visible; prior stage reads all done
    if (it + 1 < nk){
      const u32 st = (u32)((cur ^ 1) * (128*32*2));
      async_copy32(ldsA + st, gA + (it+1)*32);
      async_copy32(ldsB + st, gB + (it+1)*32);
    }
    const u16* as = &As[cur][0];
    const u16* bs = &Bs[cur][0];
    // batch all fragment loads (one ds clause, one wait), then WMMA burst
    const v16bf a0 = load_a(as + (wm + lr)*32,      lg);
    const v16bf a1 = load_a(as + (wm + 16 + lr)*32, lg);
    const v16bf b0 = load_b(bs + (wn +  0 + lr)*32, lg);
    const v16bf b1 = load_b(bs + (wn + 16 + lr)*32, lg);
    const v16bf b2 = load_b(bs + (wn + 32 + lr)*32, lg);
    const v16bf b3 = load_b(bs + (wn + 48 + lr)*32, lg);
    acc[0][0] = wmma_bf16(a0, b0, acc[0][0]);
    acc[1][0] = wmma_bf16(a1, b0, acc[1][0]);
    acc[0][1] = wmma_bf16(a0, b1, acc[0][1]);
    acc[1][1] = wmma_bf16(a1, b1, acc[1][1]);
    acc[0][2] = wmma_bf16(a0, b2, acc[0][2]);
    acc[1][2] = wmma_bf16(a1, b2, acc[1][2]);
    acc[0][3] = wmma_bf16(a0, b3, acc[0][3]);
    acc[1][3] = wmma_bf16(a1, b3, acc[1][3]);
  }

#pragma unroll
  for (int i = 0; i < 2; ++i){
#pragma unroll
    for (int j = 0; j < 4; ++j){
      const int col = nblk + wn + j*16 + lr;
      const float bv = bias ? bias[col] : 0.0f;
#pragma unroll
      for (int v = 0; v < 8; ++v){
        const int row = mblk + wm + i*16 + v + 8*lg;
        const float val = acc[i][j][v] + bv;
        if (store_f32) ((float*)Cout)[(size_t)row * N + col] = val;
        else           ((u16*)Cout)[(size_t)row * N + col]  = f2bf(val);
      }
    }
  }
}

// Flash attention: one block per (b, h, half-of-T). 8 waves; each wave owns a
// 16-row q tile, iterates causal 32-key blocks with WMMA QK^T and PV.
// qkv layout per (b,t): [h][s][d], s=0:k 1:q 2:v, row stride H_*192 bf16.
__global__ __launch_bounds__(256) void attn_wmma(
    const u16* __restrict__ qkv, u16* __restrict__ att)
{
  __shared__ u16 Vt[HS_ * T_];        // V transposed [d][key], 32 KB
  __shared__ u16 Ps[8 * 16 * 32];     // per-wave P re-layout scratch, 8 KB

  const int blk  = blockIdx.x;
  const int half = blk & 1;
  const int h    = (blk >> 1) % H_;
  const int b    = blk / (2 * H_);

  const int tid  = threadIdx.x;
  const int lane = tid & 31, wave = tid >> 5;
  const int lr   = lane & 15, lg = lane >> 4;

  // Cooperative transposed stage of V into LDS (thread t = key row t).
  {
    const uint4* src = (const uint4*)(qkv + ((size_t)(b*T_ + tid)*H_ + h)*192 + 128);
#pragma unroll
    for (int i = 0; i < 8; ++i){
      uint4 q = src[i];
      u32 w[4] = {q.x, q.y, q.z, q.w};
#pragma unroll
      for (int c = 0; c < 4; ++c){
        const int d = i*8 + c*2;
        Vt[(d    )*T_ + tid] = (u16)(w[c] & 0xFFFFu);
        Vt[(d + 1)*T_ + tid] = (u16)(w[c] >> 16);
      }
    }
  }
  __syncthreads();

  const int qrow0 = half*128 + wave*16;
  const u16* qbase = qkv + ((size_t)(b*T_ + qrow0 + lr)*H_ + h)*192 + 64; // s=1
  const v16bf aq0 = load_a(qbase,      lg);   // d 0..31
  const v16bf aq1 = load_a(qbase + 32, lg);   // d 32..63

  const u16* kbase = qkv + ((size_t)(b*T_))*H_*192 + (size_t)h*192;      // s=0
  const size_t krstride = (size_t)H_ * 192;

  float mrun[8], lsum[8];
#pragma unroll
  for (int v = 0; v < 8; ++v){ mrun[v] = -1e30f; lsum[v] = 0.0f; }
  v8f zero = {};
  v8f acc[4];
#pragma unroll
  for (int d = 0; d < 4; ++d) acc[d] = zero;

  u16* pbuf = Ps + wave * (16*32);
  const int nkb = (qrow0 + 15)/32 + 1;

  for (int jb = 0; jb < nkb; ++jb){
    const int kb = jb*32;
    const u16* kr0 = kbase + (size_t)(kb + lr)      * krstride;
    const u16* kr1 = kbase + (size_t)(kb + 16 + lr) * krstride;
    // batch the four K fragments, then the four S WMMAs
    const v16bf bk00 = load_b(kr0,      lg);
    const v16bf bk01 = load_b(kr0 + 32, lg);
    const v16bf bk10 = load_b(kr1,      lg);
    const v16bf bk11 = load_b(kr1 + 32, lg);
    v8f s0 = zero, s1 = zero;
    s0 = wmma_bf16(aq0, bk00, s0);
    s1 = wmma_bf16(aq0, bk10, s1);
    s0 = wmma_bf16(aq1, bk01, s0);
    s1 = wmma_bf16(aq1, bk11, s1);

#pragma unroll
    for (int v = 0; v < 8; ++v){
      const int row = qrow0 + v + 8*lg;
      float x0 = s0[v] * 0.125f; if (kb + lr      > row) x0 = -1e30f;
      float x1 = s1[v] * 0.125f; if (kb + 16 + lr > row) x1 = -1e30f;
      float t = fmaxf(x0, x1);
#pragma unroll
      for (int msk = 1; msk < 16; msk <<= 1) t = fmaxf(t, __shfl_xor(t, msk, 32));
      const float mn = fmaxf(mrun[v], t);
      const float p0 = __expf(x0 - mn);
      const float p1 = __expf(x1 - mn);
      float rs = p0 + p1;
#pragma unroll
      for (int msk = 1; msk < 16; msk <<= 1) rs += __shfl_xor(rs, msk, 32);
      const float alpha = __expf(mrun[v] - mn);
      lsum[v] = lsum[v]*alpha + rs;
      mrun[v] = mn;
#pragma unroll
      for (int d = 0; d < 4; ++d) acc[d][v] *= alpha;
      pbuf[(v + 8*lg)*32 + lr]      = f2bf(p0);
      pbuf[(v + 8*lg)*32 + 16 + lr] = f2bf(p1);
    }
    // wave-private LDS RAW: ensure P stores land before fragment reloads
    asm volatile("s_wait_dscnt 0" ::: "memory");
    const v16bf ap  = load_a(pbuf + lr*32, lg);
    const v16bf bv0 = load_b(Vt + ( 0 + lr)*T_ + kb, lg);
    const v16bf bv1 = load_b(Vt + (16 + lr)*T_ + kb, lg);
    const v16bf bv2 = load_b(Vt + (32 + lr)*T_ + kb, lg);
    const v16bf bv3 = load_b(Vt + (48 + lr)*T_ + kb, lg);
    acc[0] = wmma_bf16(ap, bv0, acc[0]);
    acc[1] = wmma_bf16(ap, bv1, acc[1]);
    acc[2] = wmma_bf16(ap, bv2, acc[2]);
    acc[3] = wmma_bf16(ap, bv3, acc[3]);
  }

#pragma unroll
  for (int dt = 0; dt < 4; ++dt){
#pragma unroll
    for (int v = 0; v < 8; ++v){
      const int row = qrow0 + v + 8*lg;
      const float val = acc[dt][v] / lsum[v];
      att[(size_t)(b*T_ + row)*C_ + h*HS_ + dt*16 + lr] = f2bf(val);
    }
  }
}

extern "C" void kernel_launch(void* const* d_in, const int* in_sizes, int n_in,
                              void* d_out, int out_size, void* d_ws, size_t ws_size,
                              hipStream_t stream){
  (void)in_sizes; (void)n_in; (void)out_size; (void)ws_size;
  const float* x     = (const float*)d_in[0];
  const float* Wqkv  = (const float*)d_in[1];
  const float* Wproj = (const float*)d_in[2];
  const float* bproj = (const float*)d_in[3];

  char* ws  = (char*)d_ws;
  u16* xb   = (u16*)(ws);                 // 32768*384  bf16 = 25,165,824 B
  u16* wqb  = (u16*)(ws + 25165824);      // 1152*384   bf16 =    884,736 B
  u16* wpb  = (u16*)(ws + 26050560);      // 384*384    bf16 =    294,912 B
  u16* qkvb = (u16*)(ws + 26345472);      // 32768*1152 bf16 = 75,497,472 B
  u16* attb = (u16*)(ws + 101842944);     // 32768*384  bf16 = 25,165,824 B
  u16* p1b  = (u16*)(ws + 127008768);     // 32768*384  bf16 = 25,165,824 B

  const int nx = M_*C_, nwq = F3_*C_, nwp = C_*C_;
  cvt_f32_bf16<<<(nx  + 255)/256, 256, 0, stream>>>(x,     xb,  nx);
  cvt_f32_bf16<<<(nwq + 255)/256, 256, 0, stream>>>(Wqkv,  wqb, nwq);
  cvt_f32_bf16<<<(nwp + 255)/256, 256, 0, stream>>>(Wproj, wpb, nwp);

  // qkv = x @ Wqkv^T
  gemm_bf16_wmma<<<dim3(M_/128, F3_/128), 256, 0, stream>>>(xb, wqb, nullptr, qkvb, M_, F3_, C_, 0);
  // causal attention
  attn_wmma<<<B_*H_*2, 256, 0, stream>>>(qkvb, attb);
  // projection applied twice (per reference)
  gemm_bf16_wmma<<<dim3(M_/128, C_/128), 256, 0, stream>>>(attb, wpb, bproj, p1b, M_, C_, C_, 0);
  gemm_bf16_wmma<<<dim3(M_/128, C_/128), 256, 0, stream>>>(p1b,  wpb, bproj, (float*)d_out, M_, C_, C_, 1);
}
